// PointNetPP_66529043415500
// MI455X (gfx1250) — compile-verified, matrix operand order checked by reading, not run on hardware
//
#include <hip/hip_runtime.h>
#include <math.h>
#include <stdint.h>

// ---------------------------------------------------------------------------
// PointNet++ forward for MI455X (gfx1250, wave32, WMMA + async-to-LDS)
// ---------------------------------------------------------------------------

#define G_   16
#define P_   16384
#define K_   32
#define M1_  3277
#define M2_  820
#define HID_ 128
#define LL_  16

typedef __attribute__((ext_vector_type(16))) _Float16 v16h;
typedef __attribute__((ext_vector_type(8)))  float    v8f;

#if defined(__has_builtin)
#if __has_builtin(__builtin_amdgcn_global_load_async_to_lds_b128)
#define HAVE_ASYNC_LDS 1
#else
#define HAVE_ASYNC_LDS 0
#endif
#else
#define HAVE_ASYNC_LDS 0
#endif

#if HAVE_ASYNC_LDS
// builtin signature pattern: (int4 addrspace(1)*, int4 addrspace(3)*, Ii, Ii)
typedef int v4i_gcc __attribute__((vector_size(16)));
typedef __attribute__((address_space(1))) v4i_gcc gas_v4i;
typedef __attribute__((address_space(3))) v4i_gcc las_v4i;

__device__ __forceinline__ void wait_asynccnt0()
{
#if __has_builtin(__builtin_amdgcn_s_wait_asynccnt)
    __builtin_amdgcn_s_wait_asynccnt(0);
#else
    asm volatile("s_wait_asynccnt 0x0" ::: "memory");
#endif
}
#endif

// ---------------------------------------------------------------------------
// One K-chunk (32) of the WMMA GEMM.
// A side: fast path = 4x float4 (global_load_b128) when row+chunk in bounds.
// B side: fast path = async-stage the 32x16 f32 weight tile into a per-wave
//         LDS buffer (4x global_load_async_to_lds_b128, s_wait_asynccnt 0),
//         then ds_load the strided operand values. Falls back to direct
//         global loads when async builtins are unavailable or N%16 != 0.
// A-operand layout (16-bit A 16x32): lane l holds row M=l&15,
//   K base = (l&16)?8:0, half j -> K = base + j + (j>=8 ? 8 : 0)
//   i.e. two contiguous runs [base, base+8) and [base+16, base+24).
// ---------------------------------------------------------------------------
__device__ __forceinline__
v8f gemm_chunk(const float* __restrict__ A, const float* __restrict__ W,
               int M, int KD, int N, int arow, int bcol, int n0, int kbase,
               int k0, bool vecA, bool nAligned, int lane, float* tile, v8f acc)
{
    union { v16h v; _Float16 h[16]; } ua, ub;
    const bool fullK = (k0 + 32 <= KD);

    // ---- A operand -------------------------------------------------------
    if (fullK && vecA && arow < M) {
        const float* base = A + (size_t)arow * KD + k0 + kbase;
        const float4 q0 = ((const float4*)base)[0];
        const float4 q1 = ((const float4*)base)[1];
        const float4 q2 = ((const float4*)(base + 16))[0];
        const float4 q3 = ((const float4*)(base + 16))[1];
        ua.h[0]  = (_Float16)q0.x; ua.h[1]  = (_Float16)q0.y;
        ua.h[2]  = (_Float16)q0.z; ua.h[3]  = (_Float16)q0.w;
        ua.h[4]  = (_Float16)q1.x; ua.h[5]  = (_Float16)q1.y;
        ua.h[6]  = (_Float16)q1.z; ua.h[7]  = (_Float16)q1.w;
        ua.h[8]  = (_Float16)q2.x; ua.h[9]  = (_Float16)q2.y;
        ua.h[10] = (_Float16)q2.z; ua.h[11] = (_Float16)q2.w;
        ua.h[12] = (_Float16)q3.x; ua.h[13] = (_Float16)q3.y;
        ua.h[14] = (_Float16)q3.z; ua.h[15] = (_Float16)q3.w;
    } else {
#pragma unroll
        for (int j = 0; j < 16; ++j) {
            const int kk = k0 + kbase + j + ((j >= 8) ? 8 : 0);
            float av = 0.0f;
            if (kk < KD && arow < M) av = A[(size_t)arow * KD + kk];
            ua.h[j] = (_Float16)av;
        }
    }

    // ---- B operand -------------------------------------------------------
    bool doneB = false;
#if HAVE_ASYNC_LDS
    if (fullK && nAligned) {
        // Stage W[k0..k0+32) x cols [n0..n0+16) (32 rows x 64B) into LDS.
        // 128 16B pieces; lane covers pieces lane, lane+32, lane+64, lane+96.
        const float* gsrc = W + (size_t)k0 * N + n0;
#pragma unroll
        for (int i = 0; i < 4; ++i) {
            const int piece = lane + 32 * i;
            const int row = piece >> 2;
            const int part = (piece & 3) << 2;          // float offset in row
            __builtin_amdgcn_global_load_async_to_lds_b128(
                (gas_v4i*)(gsrc + (size_t)row * N + part),
                (las_v4i*)(tile + row * 16 + part), 0, 0);
        }
        wait_asynccnt0();
#pragma unroll
        for (int j = 0; j < 16; ++j) {
            const int kr = kbase + j + ((j >= 8) ? 8 : 0);
            ub.h[j] = (_Float16)tile[kr * 16 + (lane & 15)];
        }
        doneB = true;
    }
#else
    (void)tile; (void)n0; (void)lane;
#endif
    if (!doneB) {
        if (fullK && bcol < N) {
            const float* wb = W + (size_t)(k0 + kbase) * N + bcol;
#pragma unroll
            for (int j = 0; j < 16; ++j) {
                const int koff = j + ((j >= 8) ? 8 : 0);
                ub.h[j] = (_Float16)wb[(size_t)koff * N];
            }
        } else {
#pragma unroll
            for (int j = 0; j < 16; ++j) {
                const int kk = k0 + kbase + j + ((j >= 8) ? 8 : 0);
                float bv = 0.0f;
                if (kk < KD && bcol < N) bv = W[(size_t)kk * N + bcol];
                ub.h[j] = (_Float16)bv;
            }
        }
    }

    return __builtin_amdgcn_wmma_f32_16x16x32_f16(false, ua.v, false, ub.v,
                                                  (short)0, acc, false, false);
}

// ---------------------------------------------------------------------------
// Generic WMMA GEMM:  C[M,N] = act(A[M,KD] @ W[KD,N] + bias[N])
// One wave computes one 16x16 tile of C; K-loop unrolled 2x (step 64) with
// ping-pong LDS staging buffers per wave, plus a remainder chunk.
// C/D layout: VGPR r -> M = r + ((lane&16)?8:0), N = lane&15.
// ---------------------------------------------------------------------------
__global__ __launch_bounds__(128)
void gemm_wmma_kernel(const float* __restrict__ A, const float* __restrict__ W,
                      const float* __restrict__ bias, float* __restrict__ C,
                      int M, int KD, int N, int act)
{
    __shared__ float btile[4][2][32 * 16];      // 4 waves x ping-pong x 2KB
    const int lane = threadIdx.x & 31;
    const int wave = threadIdx.x >> 5;
    const long long tile = (long long)blockIdx.x * (blockDim.x >> 5) + wave;
    const int tilesN = (N + 15) >> 4;
    const long long tm = tile / tilesN;
    const int tn = (int)(tile % tilesN);
    const int m0 = (int)(tm * 16);
    if (m0 >= M) return;                 // whole-wave uniform exit (EXEC stays full)
    const int n0 = tn * 16;
    const int lrow  = lane & 15;
    const int kbase = (lane & 16) ? 8 : 0;
    const int mhalf = (lane & 16) ? 8 : 0;
    const int arow = m0 + lrow;
    const int bcol = n0 + lrow;
    const bool vecA = ((KD & 3) == 0);   // row base 16B-aligned for float4 path
    const bool nAligned = ((N & 15) == 0);
    float* t0 = &btile[wave][0][0];
    float* t1 = &btile[wave][1][0];

    v8f acc = {};
    int k0 = 0;
    for (; k0 + 64 <= KD; k0 += 64) {
        if (k0 + 64 < KD) {              // prefetch next 64-chunk (global_prefetch_b8)
            if (arow < M) __builtin_prefetch(&A[(size_t)arow * KD + k0 + 64], 0, 1);
            if (bcol < N) __builtin_prefetch(&W[(size_t)(k0 + 64) * N + bcol], 0, 1);
        }
        acc = gemm_chunk(A, W, M, KD, N, arow, bcol, n0, kbase, k0,      vecA, nAligned, lane, t0, acc);
        acc = gemm_chunk(A, W, M, KD, N, arow, bcol, n0, kbase, k0 + 32, vecA, nAligned, lane, t1, acc);
    }
    for (; k0 < KD; k0 += 32)
        acc = gemm_chunk(A, W, M, KD, N, arow, bcol, n0, kbase, k0, vecA, nAligned, lane, t0, acc);

    union { v8f v; float f[8]; } uc; uc.v = acc;
    const float bb = (bias && bcol < N) ? bias[bcol] : 0.0f;
#pragma unroll
    for (int r = 0; r < 8; ++r) {
        const int mm = m0 + r + mhalf;
        if (mm < M && bcol < N) {
            float v = uc.f[r] + bb;
            if (act) v = fmaxf(v, 0.0f);
            C[(size_t)mm * N + bcol] = v;
        }
    }
}

// ---------------------------------------------------------------------------
// Farthest point sampling: one block per cloud, serial over m samples,
// block-wide argmax reduction in LDS. Matches reference (start idx 0, idx[s]
// emitted before the distance update).
// ---------------------------------------------------------------------------
__global__ __launch_bounds__(256)
void fps_kernel(const float* __restrict__ pos, int n, int m,
                int* __restrict__ idx, float* __restrict__ mind)
{
    const int g = blockIdx.x;
    const float* Pp = pos + (size_t)g * n * 3;
    float* md = mind + (size_t)g * n;
    int* id = idx + (size_t)g * m;
    __shared__ float sb[256];
    __shared__ int   si[256];
    __shared__ float lx, ly, lz;
    __shared__ int   lastIdx;
    const int t = threadIdx.x, nt = blockDim.x;
    for (int i = t; i < n; i += nt) md[i] = 3.0e38f;
    if (t == 0) lastIdx = 0;
    __syncthreads();
    for (int s = 0; s < m; ++s) {
        if (t == 0) {
            const int li = lastIdx;
            id[s] = li;
            lx = Pp[li * 3 + 0]; ly = Pp[li * 3 + 1]; lz = Pp[li * 3 + 2];
        }
        __syncthreads();
        const float ax = lx, ay = ly, az = lz;
        float best = -1.0f; int bi = 0x7fffffff;
        for (int i = t; i < n; i += nt) {
            const float dx = Pp[i * 3 + 0] - ax;
            const float dy = Pp[i * 3 + 1] - ay;
            const float dz = Pp[i * 3 + 2] - az;
            const float d = dx * dx + dy * dy + dz * dz;
            float mv = md[i];
            if (d < mv) { mv = d; md[i] = mv; }
            if (mv > best) { best = mv; bi = i; }
        }
        sb[t] = best; si[t] = bi;
        __syncthreads();
        for (int off = nt >> 1; off > 0; off >>= 1) {
            if (t < off) {
                if (sb[t + off] > sb[t] ||
                    (sb[t + off] == sb[t] && si[t + off] < si[t])) {
                    sb[t] = sb[t + off]; si[t] = si[t + off];
                }
            }
            __syncthreads();
        }
        if (t == 0) lastIdx = si[0];
        __syncthreads();
    }
}

// dst[g,c,:] = src[g, idx[g,c], :]
__global__ void gather_kernel(const float* __restrict__ src, const int* __restrict__ idx,
                              int n, int m, int C, float* __restrict__ dst)
{
    long long i = (long long)blockIdx.x * blockDim.x + threadIdx.x;
    const long long total = (long long)G_ * m * C;
    if (i >= total) return;
    const int ch = (int)(i % C);
    const long long gm = i / C;
    const int g = (int)(gm / m);
    const int c = (int)(gm % m);
    const int j = idx[(size_t)g * m + c];
    dst[i] = src[((size_t)g * n + j) * C + ch];
}

// ---------------------------------------------------------------------------
// Radius neighbors: one wave per center; ordered append of the first K
// in-radius points (ascending index) via wave32 ballot + popcount ranks.
// Invalid slots filled with 0; nv = number of valid slots.
// ---------------------------------------------------------------------------
__global__ __launch_bounds__(256)
void radius_kernel(const float* __restrict__ pos, const float* __restrict__ cpos,
                   int n, int m, float r2, int* __restrict__ nbr, int* __restrict__ nv)
{
    const int lane = threadIdx.x & 31;
    const long long wid = (long long)blockIdx.x * (blockDim.x >> 5) + (threadIdx.x >> 5);
    const long long total = (long long)G_ * m;
    if (wid >= total) return;
    const int g = (int)(wid / m);
    const float* Pp = pos + (size_t)g * n * 3;
    const float cx = cpos[wid * 3 + 0];
    const float cy = cpos[wid * 3 + 1];
    const float cz = cpos[wid * 3 + 2];
    int* nb = nbr + wid * K_;
    int cnt = 0;
    for (int j0 = 0; j0 < n && cnt < K_; j0 += 32) {
        const int j = j0 + lane;
        bool in = false;
        if (j < n) {
            const float dx = Pp[j * 3 + 0] - cx;
            const float dy = Pp[j * 3 + 1] - cy;
            const float dz = Pp[j * 3 + 2] - cz;
            in = (dx * dx + dy * dy + dz * dz) <= r2;
        }
        const unsigned mask = (unsigned)__ballot(in);
        if (in) {
            const int slot = cnt + __popc(mask & ((1u << lane) - 1u));
            if (slot < K_) nb[slot] = j;
        }
        cnt += __popc(mask);
    }
    if (cnt > K_) cnt = K_;
    if (lane == 0) nv[wid] = cnt;
    for (int s2 = cnt + lane; s2 < K_; s2 += 32) nb[s2] = 0;
}

// Edge features: rel = pos_j - cpos, dist = sqrt(|rel|^2 + 1e-12),
// feat = [x_j (F ch, optional) || dist]
__global__ void edge_feat_kernel(const float* __restrict__ pos, const float* __restrict__ cpos,
                                 const int* __restrict__ nbr, const float* __restrict__ x,
                                 int n, int m, int F,
                                 float* __restrict__ feat, float* __restrict__ rel)
{
    long long e = (long long)blockIdx.x * blockDim.x + threadIdx.x;
    const long long E = (long long)G_ * m * K_;
    if (e >= E) return;
    const long long cm = e / K_;
    const int g = (int)(cm / m);
    const int j = nbr[e];
    const float* pj = pos + ((size_t)g * n + j) * 3;
    const float* pc = cpos + cm * 3;
    const float rx = pj[0] - pc[0];
    const float ry = pj[1] - pc[1];
    const float rz = pj[2] - pc[2];
    rel[e * 3 + 0] = rx; rel[e * 3 + 1] = ry; rel[e * 3 + 2] = rz;
    const float dist = sqrtf(rx * rx + ry * ry + rz * rz + 1e-12f);
    float* f = feat + e * (F + 1);
    for (int q = 0; q < F; ++q) f[q] = x[((size_t)g * n + j) * F + q];
    f[F] = dist;
}

__global__ void zero_kernel(float* __restrict__ p, long long n)
{
    long long i = (long long)blockIdx.x * blockDim.x + threadIdx.x;
    if (i < n) p[i] = 0.0f;
}

// Masked BN statistics: stats[0..C) = sum, stats[C..2C) = sumsq, stats[2C] = count
__global__ void bn_stats_kernel(const float* __restrict__ H, long long M, int C,
                                const int* __restrict__ nv, float* __restrict__ stats)
{
    long long i = (long long)blockIdx.x * blockDim.x + threadIdx.x;
    if (i >= M * (long long)C) return;
    const long long row = i / C;
    const int ch = (int)(i % C);
    if (nv) {
        if ((int)(row % K_) >= nv[row / K_]) return;
    }
    const float v = H[i];
    atomicAdd(&stats[ch], v);
    atomicAdd(&stats[C + ch], v * v);
    if (ch == 0) atomicAdd(&stats[2 * C], 1.0f);
}

// In-place: H = relu(g * (H - mean) * rsqrt(var + 1e-5) + be), biased var
__global__ void bn_relu_kernel(float* __restrict__ H, long long M, int C,
                               const float* __restrict__ g, const float* __restrict__ be,
                               const float* __restrict__ stats)
{
    long long i = (long long)blockIdx.x * blockDim.x + threadIdx.x;
    if (i >= M * (long long)C) return;
    const int ch = (int)(i % C);
    const float cnt = fmaxf(stats[2 * C], 1.0f);
    const float mean = stats[ch] / cnt;
    const float var = fmaxf(stats[C + ch] / cnt - mean * mean, 0.0f);
    const float v = g[ch] * (H[i] - mean) * rsqrtf(var + 1e-5f) + be[ch];
    H[i] = fmaxf(v, 0.0f);
}

// aggr[cm,ch] = sum over valid slots of edge[cm*K+s, ch]
__global__ void aggr_x_kernel(const float* __restrict__ edge, const int* __restrict__ nv,
                              long long CM, int C, float* __restrict__ out)
{
    long long i = (long long)blockIdx.x * blockDim.x + threadIdx.x;
    if (i >= CM * (long long)C) return;
    const long long cm = i / C;
    const int ch = (int)(i % C);
    const int nvl = nv[cm];
    float s = 0.0f;
    const float* e = edge + cm * (long long)K_ * C + ch;
    for (int q = 0; q < nvl; ++q) s += e[(size_t)q * C];
    out[i] = s;
}

// newpos = cpos + (sum_s rel[s]*w[s]) / max(deg,1)
__global__ void aggr_pos_kernel(const float* __restrict__ rel, const float* __restrict__ w,
                                const int* __restrict__ nv, const float* __restrict__ cpos,
                                long long CM, float* __restrict__ outpos)
{
    long long i = (long long)blockIdx.x * blockDim.x + threadIdx.x;
    if (i >= CM * 3) return;
    const long long cm = i / 3;
    const int d = (int)(i % 3);
    const int nvl = nv[cm];
    float s = 0.0f;
    for (int q = 0; q < nvl; ++q) s += rel[(cm * K_ + q) * 3 + d] * w[cm * K_ + q];
    const float deg = (float)(nvl > 0 ? nvl : 1);
    outpos[i] = cpos[i] + s / deg;
}

// out[row] = [A(rowA) || B(row)]; aGroupRows>0 => A broadcast (rowA = row/aGroupRows)
__global__ void concat_kernel(const float* __restrict__ A, int Ca, int aGroupRows,
                              const float* __restrict__ B, int Cb,
                              long long Mrows, float* __restrict__ out)
{
    long long i = (long long)blockIdx.x * blockDim.x + threadIdx.x;
    const int Ct = Ca + Cb;
    if (i >= Mrows * (long long)Ct) return;
    const long long row = i / Ct;
    const int ch = (int)(i % Ct);
    if (ch < Ca) {
        const long long ar = aGroupRows ? (row / aGroupRows) : row;
        out[i] = A[ar * (long long)Ca + ch];
    } else {
        out[i] = B[row * (long long)Cb + (ch - Ca)];
    }
}

__global__ void maxpool_kernel(const float* __restrict__ X, int rows, int C,
                               float* __restrict__ out)
{
    const int g = blockIdx.x;
    const int ch = threadIdx.x;
    if (ch >= C) return;
    float m = -3.0e38f;
    for (int r = 0; r < rows; ++r) m = fmaxf(m, X[((size_t)g * rows + r) * C + ch]);
    out[(size_t)g * C + ch] = m;
}

// 16-layer LSTM over the G=16 sequence; one thread per hidden unit.
__global__ __launch_bounds__(HID_)
void lstm_kernel(const float* __restrict__ gx, const float* __restrict__ Wih,
                 const float* __restrict__ Whh, const float* __restrict__ bih,
                 const float* __restrict__ bhh, float* __restrict__ out)
{
    __shared__ float xbuf[G_][HID_];
    __shared__ float hs[HID_];
    const int t = threadIdx.x;
    for (int i = t; i < G_ * HID_; i += HID_) (&xbuf[0][0])[i] = gx[i];
    __syncthreads();
    for (int l = 0; l < LL_; ++l) {
        const float* Wi = Wih + (size_t)l * 4 * HID_ * HID_;
        const float* Wh = Whh + (size_t)l * 4 * HID_ * HID_;
        const float* bi = bih + (size_t)l * 4 * HID_;
        const float* bh = bhh + (size_t)l * 4 * HID_;
        float c = 0.0f;
        hs[t] = 0.0f;
        __syncthreads();
        for (int s = 0; s < G_; ++s) {
            float gi = bi[0 * HID_ + t] + bh[0 * HID_ + t];
            float gf = bi[1 * HID_ + t] + bh[1 * HID_ + t];
            float gg = bi[2 * HID_ + t] + bh[2 * HID_ + t];
            float go = bi[3 * HID_ + t] + bh[3 * HID_ + t];
            for (int q = 0; q < HID_; ++q) {
                const float xv = xbuf[s][q];
                const float hv = hs[q];
                gi += Wi[(size_t)(0 * HID_ + t) * HID_ + q] * xv + Wh[(size_t)(0 * HID_ + t) * HID_ + q] * hv;
                gf += Wi[(size_t)(1 * HID_ + t) * HID_ + q] * xv + Wh[(size_t)(1 * HID_ + t) * HID_ + q] * hv;
                gg += Wi[(size_t)(2 * HID_ + t) * HID_ + q] * xv + Wh[(size_t)(2 * HID_ + t) * HID_ + q] * hv;
                go += Wi[(size_t)(3 * HID_ + t) * HID_ + q] * xv + Wh[(size_t)(3 * HID_ + t) * HID_ + q] * hv;
            }
            __syncthreads();
            const float si = 1.0f / (1.0f + expf(-gi));
            const float sf = 1.0f / (1.0f + expf(-gf));
            const float so = 1.0f / (1.0f + expf(-go));
            c = sf * c + si * tanhf(gg);
            const float hv = so * tanhf(c);
            hs[t] = hv;
            xbuf[s][t] = hv;
            __syncthreads();
        }
    }
    for (int i = t; i < G_ * HID_; i += HID_) out[i] = (&xbuf[0][0])[i];
}

// knn(k=3) interpolate with 1/d^2 weights (PyG knn_interpolate)
__global__ void knn3_kernel(const float* __restrict__ xsrc, const float* __restrict__ psrc,
                            const float* __restrict__ ptgt, int Ms, int Mt, int C,
                            float* __restrict__ out)
{
    long long i = (long long)blockIdx.x * blockDim.x + threadIdx.x;
    const long long total = (long long)G_ * Mt;
    if (i >= total) return;
    const int g = (int)(i / Mt);
    const float tx = ptgt[i * 3 + 0], ty = ptgt[i * 3 + 1], tz = ptgt[i * 3 + 2];
    const float* Ps = psrc + (size_t)g * Ms * 3;
    float d0 = 3.0e38f, d1 = 3.0e38f, d2 = 3.0e38f;
    int i0 = 0, i1 = 0, i2 = 0;
    for (int j = 0; j < Ms; ++j) {
        const float dx = Ps[j * 3 + 0] - tx;
        const float dy = Ps[j * 3 + 1] - ty;
        const float dz = Ps[j * 3 + 2] - tz;
        const float d = dx * dx + dy * dy + dz * dz;
        if (d < d0)      { d2 = d1; i2 = i1; d1 = d0; i1 = i0; d0 = d; i0 = j; }
        else if (d < d1) { d2 = d1; i2 = i1; d1 = d; i1 = j; }
        else if (d < d2) { d2 = d; i2 = j; }
    }
    const float w0 = 1.0f / fmaxf(d0, 1e-16f);
    const float w1 = 1.0f / fmaxf(d1, 1e-16f);
    const float w2 = 1.0f / fmaxf(d2, 1e-16f);
    const float ws = w0 + w1 + w2;
    const float* Xs = xsrc + (size_t)g * Ms * C;
    float* o = out + i * (long long)C;
    for (int ch = 0; ch < C; ++ch)
        o[ch] = (w0 * Xs[(size_t)i0 * C + ch] + w1 * Xs[(size_t)i1 * C + ch] +
                 w2 * Xs[(size_t)i2 * C + ch]) / ws;
}

// ---------------------------------------------------------------------------
// Host orchestration
// ---------------------------------------------------------------------------

struct MLP2 { const float *W1, *b1, *g, *be, *W2, *b2; };
struct Lin  { const float *W, *b; };

#define LAUNCH_N(kern, n, ...) do {                                        \
    long long _n = (long long)(n);                                         \
    if (_n > 0) {                                                          \
        unsigned _g = (unsigned)((_n + 255) / 256);                        \
        kern<<<dim3(_g), dim3(256), 0, stream>>>(__VA_ARGS__);             \
    } } while (0)

static void run_gemm(const float* A, const float* W, const float* b, float* C,
                     long long M, int KD, int N, int act, hipStream_t stream)
{
    const long long tiles = ((M + 15) / 16) * (long long)((N + 15) / 16);
    const unsigned blocks = (unsigned)((tiles + 3) / 4);
    gemm_wmma_kernel<<<dim3(blocks), dim3(128), 0, stream>>>(A, W, b, C, (int)M, KD, N, act);
}

static void run_mlp2(const float* A, long long M, int d0, int d1, int d2,
                     const MLP2& p, const int* nv, float* H, float* Out,
                     float* stats, hipStream_t stream)
{
    run_gemm(A, p.W1, p.b1, H, M, d0, d1, 0, stream);
    LAUNCH_N(zero_kernel, 2 * d1 + 1, stats, (long long)(2 * d1 + 1));
    LAUNCH_N(bn_stats_kernel, M * (long long)d1, H, M, d1, nv, stats);
    LAUNCH_N(bn_relu_kernel, M * (long long)d1, H, M, d1, p.g, p.be, stats);
    run_gemm(H, p.W2, p.b2, Out, M, d1, d2, 0, stream);
}

extern "C" void kernel_launch(void* const* d_in, const int* in_sizes, int n_in,
                              void* d_out, int out_size, void* d_ws, size_t ws_size,
                              hipStream_t stream)
{
    (void)n_in; (void)out_size; (void)ws_size;

    // ---- map inputs (handles insertion-order and JAX sorted-key flattening)
    const float* pos = nullptr;
    MLP2 s1x, s1p, s1u, s2x, s2p, s2u, sa3m;
    Lin fp3, fp2, fp1, hl1, hl2;
    const float *Wih = nullptr, *Whh = nullptr, *bih = nullptr, *bhh = nullptr;

    int ii = 0;
    auto F = [&]() { return (const float*)d_in[ii++]; };
    if (in_sizes[0] == G_ * P_ * 3) {
        // insertion order: pos, sa1, sa2, sa3, lstm, fp3, fp2, fp1, head
        pos = F();
        auto mlp = [&](MLP2& m) { m.W1 = F(); m.b1 = F(); m.g = F(); m.be = F(); m.W2 = F(); m.b2 = F(); };
        mlp(s1x); mlp(s1p); mlp(s1u);
        mlp(s2x); mlp(s2p); mlp(s2u);
        mlp(sa3m);
        Wih = F(); Whh = F(); bih = F(); bhh = F();
        fp3.W = F(); fp3.b = F(); fp2.W = F(); fp2.b = F(); fp1.W = F(); fp1.b = F();
        hl1.W = F(); hl1.b = F(); hl2.W = F(); hl2.b = F();
    } else {
        // JAX sorted keys: params(fp1,fp2,fp3,head,lstm{Whh,Wih,bhh,bih},
        //                  sa1{msg_pos,msg_x,upd}{be,g,l1,l2}, sa2, sa3), pos
        auto mlps = [&](MLP2& m) { m.be = F(); m.g = F(); m.W1 = F(); m.b1 = F(); m.W2 = F(); m.b2 = F(); };
        fp1.W = F(); fp1.b = F(); fp2.W = F(); fp2.b = F(); fp3.W = F(); fp3.b = F();
        hl1.W = F(); hl1.b = F(); hl2.W = F(); hl2.b = F();
        Whh = F(); Wih = F(); bhh = F(); bih = F();
        mlps(s1p); mlps(s1x); mlps(s1u);
        mlps(s2p); mlps(s2x); mlps(s2u);
        mlps(sa3m);
        pos = F();
    }

    // ---- workspace bump allocator (256B aligned)
    char* bp = (char*)d_ws;
    auto AF = [&](size_t n) { float* r = (float*)bp; bp += ((n * 4 + 255) / 256) * 256; return r; };
    auto AI = [&](size_t n) { int*   r = (int*)bp;   bp += ((n * 4 + 255) / 256) * 256; return r; };

    const long long E1 = (long long)G_ * M1_ * K_;   // 1,677,824 edges
    const long long E2 = (long long)G_ * M2_ * K_;   //   419,840 edges
    const size_t BIG = (size_t)E1 * 32;              // >= E2*64 and G*P*128
    float* BIGA = AF(BIG);                           // H1 / Hp1 / feat2 / Hp2 / f1
    float* BIGB = AF(BIG);                           // edge1 / edge2 / hh
    float* BIGC = AF((size_t)E2 * 33);               // H2 / f2i / f1i

    float* mind   = AF((size_t)G_ * P_);
    int*   idx1   = AI((size_t)G_ * M1_);
    float* cpos1  = AF((size_t)G_ * M1_ * 3);
    int*   nbr1   = AI((size_t)E1);
    int*   nv1    = AI((size_t)G_ * M1_);
    float* feat1  = AF((size_t)E1);
    float* rel1   = AF((size_t)E1 * 3);
    float* w1     = AF((size_t)E1);
    float* aggrx1 = AF((size_t)G_ * M1_ * 32);
    float* x1     = AF((size_t)G_ * M1_ * 32);
    float* pos1   = AF((size_t)G_ * M1_ * 3);
    float* Hu     = AF((size_t)G_ * M1_ * 32);       // >= G*M2*64 too
    int*   idx2   = AI((size_t)G_ * M2_);
    float* cpos2  = AF((size_t)G_ * M2_ * 3);
    int*   nbr2   = AI((size_t)E2);
    int*   nv2    = AI((size_t)G_ * M2_);
    float* xdest2 = AF((size_t)G_ * M2_ * 32);
    float* rel2   = AF((size_t)E2 * 3);
    float* w2     = AF((size_t)E2);
    float* aggrx2 = AF((size_t)G_ * M2_ * 64);
    float* updin2 = AF((size_t)G_ * M2_ * 96);
    float* x2     = AF((size_t)G_ * M2_ * 64);
    float* pos2   = AF((size_t)G_ * M2_ * 3);
    float* x3     = AF((size_t)G_ * M2_ * 128);
    float* gxb    = AF((size_t)G_ * HID_);
    float* hseq   = AF((size_t)G_ * HID_);
    float* f3in   = AF((size_t)G_ * M2_ * 192);
    float* f3     = AF((size_t)G_ * M2_ * 64);
    float* f2in   = AF((size_t)G_ * M1_ * 96);
    float* f2     = AF((size_t)G_ * M1_ * 32);
    float* stats  = AF(512);

    const long long CM1 = (long long)G_ * M1_;
    const long long CM2 = (long long)G_ * M2_;

    // ======================= SA1 =======================
    fps_kernel<<<dim3(G_), dim3(256), 0, stream>>>(pos, P_, M1_, idx1, mind);
    LAUNCH_N(gather_kernel, CM1 * 3, pos, idx1, P_, M1_, 3, cpos1);
    {
        const unsigned blocks = (unsigned)((CM1 + 7) / 8);
        radius_kernel<<<dim3(blocks), dim3(256), 0, stream>>>(pos, cpos1, P_, M1_, 0.2f * 0.2f, nbr1, nv1);
    }
    LAUNCH_N(edge_feat_kernel, E1, pos, cpos1, nbr1, (const float*)nullptr, P_, M1_, 0, feat1, rel1);
    run_mlp2(feat1, E1, 1, 32, 32, s1x, nv1, BIGA, BIGB /*edge1*/, stats, stream);   // msg_x
    run_mlp2(BIGB, E1, 32, 32, 1, s1p, nv1, BIGA, w1, stats, stream);                // msg_pos
    LAUNCH_N(aggr_x_kernel, CM1 * 32, BIGB, nv1, CM1, 32, aggrx1);
    LAUNCH_N(aggr_pos_kernel, CM1 * 3, rel1, w1, nv1, cpos1, CM1, pos1);
    run_mlp2(aggrx1, CM1, 32, 32, 32, s1u, (const int*)nullptr, Hu, x1, stats, stream); // upd

    // ======================= SA2 =======================
    fps_kernel<<<dim3(G_), dim3(256), 0, stream>>>(pos1, M1_, M2_, idx2, mind);
    LAUNCH_N(gather_kernel, CM2 * 3, pos1, idx2, M1_, M2_, 3, cpos2);
    LAUNCH_N(gather_kernel, CM2 * 32, x1, idx2, M1_, M2_, 32, xdest2);
    {
        const unsigned blocks = (unsigned)((CM2 + 7) / 8);
        radius_kernel<<<dim3(blocks), dim3(256), 0, stream>>>(pos1, cpos2, M1_, M2_, 0.4f * 0.4f, nbr2, nv2);
    }
    LAUNCH_N(edge_feat_kernel, E2, pos1, cpos2, nbr2, x1, M1_, M2_, 32, BIGA /*feat2*/, rel2);
    run_mlp2(BIGA, E2, 33, 32, 64, s2x, nv2, BIGC /*H2*/, BIGB /*edge2*/, stats, stream); // msg_x
    run_mlp2(BIGB, E2, 64, 64, 1, s2p, nv2, BIGA /*Hp2*/, w2, stats, stream);             // msg_pos
    LAUNCH_N(aggr_x_kernel, CM2 * 64, BIGB, nv2, CM2, 64, aggrx2);
    LAUNCH_N(aggr_pos_kernel, CM2 * 3, rel2, w2, nv2, cpos2, CM2, pos2);
    LAUNCH_N(concat_kernel, CM2 * 96, xdest2, 32, 0, aggrx2, 64, CM2, updin2);
    run_mlp2(updin2, CM2, 96, 64, 64, s2u, (const int*)nullptr, Hu, x2, stats, stream);   // upd

    // ================== SA3 + pool + LSTM ==================
    run_mlp2(x2, CM2, 64, 64, 128, sa3m, (const int*)nullptr, Hu, x3, stats, stream);
    maxpool_kernel<<<dim3(G_), dim3(HID_), 0, stream>>>(x3, M2_, 128, gxb);
    lstm_kernel<<<dim3(1), dim3(HID_), 0, stream>>>(gxb, Wih, Whh, bih, bhh, hseq);

    // ================== FP3 / FP2 / FP1 / head ==================
    LAUNCH_N(concat_kernel, CM2 * 192, hseq, 128, M2_, x2, 64, CM2, f3in);
    run_gemm(f3in, fp3.W, fp3.b, f3, CM2, 192, 64, 0, stream);

    LAUNCH_N(knn3_kernel, CM1, f3, pos2, pos1, M2_, M1_, 64, BIGC /*f2i*/);
    LAUNCH_N(concat_kernel, CM1 * 96, BIGC, 64, 0, x1, 32, CM1, f2in);
    run_gemm(f2in, fp2.W, fp2.b, f2, CM1, 96, 32, 0, stream);

    const long long GP = (long long)G_ * P_;
    LAUNCH_N(knn3_kernel, GP, f2, pos1, pos, M1_, P_, 32, BIGC /*f1i*/);
    run_gemm(BIGC, fp1.W, fp1.b, BIGA /*f1*/, GP, 32, 128, 0, stream);
    run_gemm(BIGA, hl1.W, hl1.b, BIGB /*hh*/, GP, 128, 64, 1, stream);
    run_gemm(BIGB, hl2.W, hl2.b, (float*)d_out, GP, 64, 3, 0, stream);
}
